// MoE_730144440331
// MI455X (gfx1250) — compile-verified, matrix operand order checked by reading, not run on hardware
//
#include <hip/hip_runtime.h>

#define T_TOK 8192
#define E_DIM 1024
#define FF_DIM 4096
#define NE 8
#define TOPK 2
#define CAPACITY 2048
#define RSLOT (T_TOK * TOPK)

typedef __bf16 bf16_t;
typedef __bf16 v16bf __attribute__((ext_vector_type(16)));
typedef __bf16 v8bf  __attribute__((ext_vector_type(8)));
typedef __bf16 v4bf  __attribute__((ext_vector_type(4)));
typedef float  v8f   __attribute__((ext_vector_type(8)));
typedef int    v4i32 __attribute__((vector_size(16)));

// gfx1250 async global->LDS path (ASYNCcnt-tracked), guarded so the file still
// compiles if this toolchain doesn't declare the builtins.
#if defined(__has_builtin)
#if __has_builtin(__builtin_amdgcn_global_load_async_to_lds_b128) && \
    __has_builtin(__builtin_amdgcn_s_wait_asynccnt)
#define USE_ASYNC_LDS 1
#endif
#endif
#ifndef USE_ASYNC_LDS
#define USE_ASYNC_LDS 0
#endif

// builtin signature (from hipcc diagnostic): (as1 v4i32*, as3 v4i32*, Ii, Ii)
#define ASYNC_GPTR(p) ((__attribute__((address_space(1))) v4i32*)(const void*)(p))
#define ASYNC_LPTR(p) ((__attribute__((address_space(3))) v4i32*)(void*)(p))

// ---------------------------------------------------------------------------
// Kernel 1: router — one wave32 per token. Dot with 8 router rows, softmax,
// top-2 (lowest-index tie break like jax.lax.top_k), L1-normalize the pair.
// Per-block importance partials via fixed-order reduction (deterministic).
// ---------------------------------------------------------------------------
__global__ void __launch_bounds__(256)
moe_router_kernel(const float* __restrict__ x, const float* __restrict__ rw,
                  const float* __restrict__ rb, int* __restrict__ eidx,
                  float* __restrict__ eval, float* __restrict__ imp_part) {
  const int wave = threadIdx.x >> 5;
  const int lane = threadIdx.x & 31;
  const int t = blockIdx.x * 8 + wave;

  float acc[NE];
#pragma unroll
  for (int n = 0; n < NE; ++n) acc[n] = 0.f;

  const float* xr = x + (size_t)t * E_DIM;
  for (int e = lane; e < E_DIM; e += 32) {
    const float xv = xr[e];
#pragma unroll
    for (int n = 0; n < NE; ++n) acc[n] = fmaf(xv, rw[n * E_DIM + e], acc[n]);
  }
#pragma unroll
  for (int n = 0; n < NE; ++n) {
#pragma unroll
    for (int off = 16; off >= 1; off >>= 1) acc[n] += __shfl_xor(acc[n], off, 32);
  }

  __shared__ float simp[8][NE];
  if (lane == 0) {
    float p[NE];
    float m = -1e30f;
#pragma unroll
    for (int n = 0; n < NE; ++n) { p[n] = acc[n] + rb[n]; m = fmaxf(m, p[n]); }
    float s = 0.f;
#pragma unroll
    for (int n = 0; n < NE; ++n) { p[n] = __expf(p[n] - m); s += p[n]; }
    const float inv = 1.f / s;
#pragma unroll
    for (int n = 0; n < NE; ++n) p[n] *= inv;

    int i1 = 0;
#pragma unroll
    for (int n = 1; n < NE; ++n) if (p[n] > p[i1]) i1 = n;
    int i2 = (i1 == 0) ? 1 : 0;
#pragma unroll
    for (int n = 0; n < NE; ++n) if (n != i1 && p[n] > p[i2]) i2 = n;

    const float vsum = p[i1] + p[i2];
    eidx[t * 2 + 0] = i1;
    eidx[t * 2 + 1] = i2;
    eval[t * 2 + 0] = p[i1] / vsum;
    eval[t * 2 + 1] = p[i2] / vsum;
#pragma unroll
    for (int n = 0; n < NE; ++n) simp[wave][n] = p[n];
  }
  __syncthreads();
  if (threadIdx.x < NE) {
    float s = 0.f;
#pragma unroll
    for (int w = 0; w < 8; ++w) s += simp[w][threadIdx.x];
    imp_part[blockIdx.x * NE + threadIdx.x] = s;
  }
}

// ---------------------------------------------------------------------------
// Kernel 2: single-block exclusive scan over per-thread expert histograms.
// Ranks in original slot order == reference's stable-sort-by-expert rank.
// Also emits loss_aux to d_out[T*E].
// ---------------------------------------------------------------------------
__global__ void __launch_bounds__(1024)
moe_scan_kernel(const int* __restrict__ eidx, int* __restrict__ ranks,
                const float* __restrict__ imp_part, float* __restrict__ loss_out) {
  __shared__ int hist[1024 * NE];
  const int tid = threadIdx.x;
  const int base = tid * 16;

  int le[16];
  int cnt[NE];
#pragma unroll
  for (int n = 0; n < NE; ++n) cnt[n] = 0;
#pragma unroll
  for (int i = 0; i < 16; ++i) { le[i] = eidx[base + i]; cnt[le[i]]++; }
#pragma unroll
  for (int n = 0; n < NE; ++n) hist[tid * NE + n] = cnt[n];
  __syncthreads();

  for (int off = 1; off < 1024; off <<= 1) {
    int v[NE];
    if (tid >= off) {
#pragma unroll
      for (int n = 0; n < NE; ++n) v[n] = hist[(tid - off) * NE + n];
    }
    __syncthreads();
    if (tid >= off) {
#pragma unroll
      for (int n = 0; n < NE; ++n) hist[tid * NE + n] += v[n];
    }
    __syncthreads();
  }

  int ex[NE];
#pragma unroll
  for (int n = 0; n < NE; ++n) ex[n] = tid ? hist[(tid - 1) * NE + n] : 0;
#pragma unroll
  for (int i = 0; i < 16; ++i) { const int n = le[i]; ranks[base + i] = ex[n]; ex[n]++; }

  if (tid == 0) {
    float loss = 0.f;
    for (int n = 0; n < NE; ++n) {
      float imp = 0.f;
      for (int b = 0; b < 1024; ++b) imp += imp_part[b * NE + n];
      imp *= (1.f / (float)T_TOK);
      const float load = (float)hist[1023 * NE + n] / (float)RSLOT;
      loss += load * imp;
    }
    loss_out[0] = (float)NE * loss;
  }
}

// ---------------------------------------------------------------------------
// Kernel 3: dispatch rows (rank < CAP) into bf16 expert buffers.
// ---------------------------------------------------------------------------
__global__ void __launch_bounds__(256)
moe_dispatch_kernel(const float* __restrict__ x, const int* __restrict__ eidx,
                    const int* __restrict__ ranks, bf16_t* __restrict__ xd) {
  const int r = blockIdx.x;
  const int rank = ranks[r];
  if (rank >= CAPACITY) return;
  const int n = eidx[r];
  const int src = r >> 1;

  const float4* xin = (const float4*)(x + (size_t)src * E_DIM);
  v4bf* out = (v4bf*)(xd + ((size_t)(n * CAPACITY + rank)) * E_DIM);
  const int i = threadIdx.x;  // 256 threads x float4 == 1024 elems
  const float4 v = xin[i];
  v4bf o;
  o[0] = (bf16_t)v.x; o[1] = (bf16_t)v.y; o[2] = (bf16_t)v.z; o[3] = (bf16_t)v.w;
  out[i] = o;
}

// ---------------------------------------------------------------------------
// WMMA fragment loader: 16-bit A/B layout (ISA 7.12.2). Lane l (0..31):
// row/col = l&15; elems 0..7 = K (l>>4)*8..+7 ; elems 8..15 = K 16+(l>>4)*8.
// LDS rows padded to 40 bf16 (80B = 20 banks) -> conflict-free b128 reads.
// ---------------------------------------------------------------------------
__device__ __forceinline__ v16bf load_frag_b16(const bf16_t* p, int half) {
  const v8bf lo = *(const v8bf*)(p + half * 8);
  const v8bf hi = *(const v8bf*)(p + 16 + half * 8);
  v16bf f;
#pragma unroll
  for (int i = 0; i < 8; ++i) { f[i] = lo[i]; f[i + 8] = hi[i]; }
  return f;
}

// ---------------------------------------------------------------------------
// Kernels 4/5: grouped GEMM, double-buffered LDS pipeline.
// Block = 128(M)x128(N), 8 waves, wave = 64x32 = 4x2 wmma_f32_16x16x32_bf16
// fragments, K-step 32. A tile (already bf16) is staged with the gfx1250
// async global->LDS path when available (ASYNCcnt); B tiles are fetched to
// registers, converted fp32->bf16, and ds_stored — overlapped with the WMMA
// block of the current K-step. One barrier per K-step.
// GATED: D = bf16( silu(A@B0^T) * (A@B1^T) )   (gate+enc fused, 16 wmma/step)
// else : D = f32 ( A@B0^T )                    (decoder)
// Per-expert fp32 weight panel (<=64MB) stays resident in the 192MB L2 across
// the tileM sweep, so HBM reads each weight byte once per expert.
// ---------------------------------------------------------------------------
template <bool GATED>
__global__ void __launch_bounds__(256)
moe_gemm_kernel(const bf16_t* __restrict__ A, const float* __restrict__ B0,
                const float* __restrict__ B1, bf16_t* __restrict__ Dh,
                float* __restrict__ Df, int M, int Ndim, int Kdim) {
  constexpr int LD = 40;  // padded LDS row stride (bf16 elems)
  __shared__ __align__(16) bf16_t ldsA[2][128 * LD];
  __shared__ __align__(16) bf16_t ldsB0[2][128 * LD];
  __shared__ __align__(16) bf16_t ldsB1[2][128 * LD];

  const int n = blockIdx.z;
  const int tileN = blockIdx.x * 128;
  const int tileM = blockIdx.y * 128;

  const bf16_t* Ab = A + (size_t)n * M * Kdim;
  const float* B0b = B0 + (size_t)n * Ndim * Kdim;
  const float* B1b = GATED ? (B1 + (size_t)n * Ndim * Kdim) : nullptr;

  const int tid = threadIdx.x;
  const int lane = tid & 31;
  const int wid = tid >> 5;
  const int waveM = wid >> 2;  // 0..1
  const int waveN = wid & 3;   // 0..3
  const int rowInF = lane & 15;
  const int half = lane >> 4;

  // staging coordinates: A = 2x16B chunks/thread, B = 16 f32 cols/thread
  const int aRow = tid >> 1;
  const int aSub = (tid & 1) * 2;  // chunks aSub, aSub+1 (each 8 bf16)
  const int bRow = tid >> 1;
  const int bHalf = tid & 1;

  // ---- stage A tile (bf16): async global->LDS when available ----
  auto stageA = [&](int k0, bf16_t* buf) {
#pragma unroll
    for (int j = 0; j < 2; ++j) {
      const bf16_t* g = Ab + (size_t)(tileM + aRow) * Kdim + k0 + (aSub + j) * 8;
      bf16_t* l = &buf[aRow * LD + (aSub + j) * 8];
#if USE_ASYNC_LDS
      __builtin_amdgcn_global_load_async_to_lds_b128(ASYNC_GPTR(g), ASYNC_LPTR(l), 0, 0);
#else
      *(uint4*)l = *(const uint4*)g;
#endif
    }
  };
  // ---- fetch B tiles (fp32) into registers ----
  auto fetchB = [&](int k0, float4* r0, float4* r1) {
    const float* s0 = B0b + (size_t)(tileN + bRow) * Kdim + k0 + bHalf * 16;
#pragma unroll
    for (int j = 0; j < 4; ++j) r0[j] = *(const float4*)(s0 + j * 4);
    if (GATED) {
      const float* s1 = B1b + (size_t)(tileN + bRow) * Kdim + k0 + bHalf * 16;
#pragma unroll
      for (int j = 0; j < 4; ++j) r1[j] = *(const float4*)(s1 + j * 4);
    }
  };
  // ---- convert fp32->bf16 and ds_store B tiles ----
  auto storeB = [&](bf16_t* d0, bf16_t* d1, const float4* r0, const float4* r1) {
#pragma unroll
    for (int j = 0; j < 4; ++j) {
      v4bf o;
      o[0] = (bf16_t)r0[j].x; o[1] = (bf16_t)r0[j].y;
      o[2] = (bf16_t)r0[j].z; o[3] = (bf16_t)r0[j].w;
      *(v4bf*)&d0[bRow * LD + bHalf * 16 + j * 4] = o;
    }
    if (GATED) {
#pragma unroll
      for (int j = 0; j < 4; ++j) {
        v4bf o;
        o[0] = (bf16_t)r1[j].x; o[1] = (bf16_t)r1[j].y;
        o[2] = (bf16_t)r1[j].z; o[3] = (bf16_t)r1[j].w;
        *(v4bf*)&d1[bRow * LD + bHalf * 16 + j * 4] = o;
      }
    }
  };

  v8f acc0[4][2];
  v8f acc1[4][2];
#pragma unroll
  for (int mi = 0; mi < 4; ++mi)
#pragma unroll
    for (int ni = 0; ni < 2; ++ni)
#pragma unroll
      for (int v = 0; v < 8; ++v) {
        acc0[mi][ni][v] = 0.f;
        if (GATED) acc1[mi][ni][v] = 0.f;
      }

  float4 b0r[4], b1r[4];

  // ---- pipeline prologue: stage K-panel 0 into buffer 0 ----
  stageA(0, ldsA[0]);
  fetchB(0, b0r, b1r);
  storeB(ldsB0[0], ldsB1[0], b0r, b1r);
#if USE_ASYNC_LDS
  __builtin_amdgcn_s_wait_asynccnt(0);
#endif
  __syncthreads();

  int cur = 0;
  for (int k0 = 0; k0 < Kdim; k0 += 32) {
    const bool has_next = (k0 + 32) < Kdim;
    if (has_next) {
      stageA(k0 + 32, ldsA[cur ^ 1]);   // async A into the other buffer
      fetchB(k0 + 32, b0r, b1r);        // B panel into registers
      __builtin_prefetch(B0b + (size_t)(tileN + bRow) * Kdim + k0 + 64, 0, 1);
    }

    // ---- compute current K-panel ----
    const bf16_t* cA = ldsA[cur];
    const bf16_t* cB0 = ldsB0[cur];
    const bf16_t* cB1 = ldsB1[cur];
    v16bf a[4], b0[2], b1[2];
#pragma unroll
    for (int mi = 0; mi < 4; ++mi)
      a[mi] = load_frag_b16(&cA[(waveM * 64 + mi * 16 + rowInF) * LD], half);
#pragma unroll
    for (int ni = 0; ni < 2; ++ni) {
      b0[ni] = load_frag_b16(&cB0[(waveN * 32 + ni * 16 + rowInF) * LD], half);
      if (GATED) b1[ni] = load_frag_b16(&cB1[(waveN * 32 + ni * 16 + rowInF) * LD], half);
    }
#pragma unroll
    for (int mi = 0; mi < 4; ++mi)
#pragma unroll
      for (int ni = 0; ni < 2; ++ni) {
        acc0[mi][ni] = __builtin_amdgcn_wmma_f32_16x16x32_bf16(
            false, a[mi], false, b0[ni], (short)0, acc0[mi][ni], false, false);
        if (GATED)
          acc1[mi][ni] = __builtin_amdgcn_wmma_f32_16x16x32_bf16(
              false, a[mi], false, b1[ni], (short)0, acc1[mi][ni], false, false);
      }

    if (has_next) {
      storeB(ldsB0[cur ^ 1], ldsB1[cur ^ 1], b0r, b1r);
#if USE_ASYNC_LDS
      __builtin_amdgcn_s_wait_asynccnt(0);
#endif
    }
    __syncthreads();
    cur ^= 1;
  }

  // ---- epilogue: C/D layout — elem v: row = v + 8*half, col = lane&15 ----
#pragma unroll
  for (int mi = 0; mi < 4; ++mi)
#pragma unroll
    for (int ni = 0; ni < 2; ++ni) {
      const int row0 = tileM + waveM * 64 + mi * 16 + half * 8;
      const int col = tileN + waveN * 32 + ni * 16 + rowInF;
#pragma unroll
      for (int v = 0; v < 8; ++v) {
        const size_t idx = (size_t)n * M * Ndim + (size_t)(row0 + v) * Ndim + col;
        if (GATED) {
          const float g = acc0[mi][ni][v];
          const float e = acc1[mi][ni][v];
          const float s = g / (1.f + __expf(-g));  // silu
          Dh[idx] = (bf16_t)(s * e);
        } else {
          Df[idx] = acc0[mi][ni][v];
        }
      }
    }
}

// ---------------------------------------------------------------------------
// Kernel 6: combine — each token gathers its 2 expert slots (zero if dropped),
// weights and sums. Deterministic (fixed-order add, no atomics).
// ---------------------------------------------------------------------------
__global__ void __launch_bounds__(256)
moe_combine_kernel(const float* __restrict__ out2, const int* __restrict__ eidx,
                   const int* __restrict__ ranks, const float* __restrict__ eval,
                   float* __restrict__ out) {
  const int t = blockIdx.x;
  const int n0 = eidx[t * 2 + 0], n1 = eidx[t * 2 + 1];
  const int r0 = ranks[t * 2 + 0], r1 = ranks[t * 2 + 1];
  const float v0 = eval[t * 2 + 0], v1 = eval[t * 2 + 1];
  const bool u0 = (r0 < CAPACITY), u1 = (r1 < CAPACITY);
  const int i = threadIdx.x;

  float4 acc = make_float4(0.f, 0.f, 0.f, 0.f);
  if (u0) {
    const float4 a = ((const float4*)(out2 + ((size_t)(n0 * CAPACITY + r0)) * E_DIM))[i];
    acc.x += v0 * a.x; acc.y += v0 * a.y; acc.z += v0 * a.z; acc.w += v0 * a.w;
  }
  if (u1) {
    const float4 a = ((const float4*)(out2 + ((size_t)(n1 * CAPACITY + r1)) * E_DIM))[i];
    acc.x += v1 * a.x; acc.y += v1 * a.y; acc.z += v1 * a.z; acc.w += v1 * a.w;
  }
  ((float4*)(out + (size_t)t * E_DIM))[i] = acc;
}

// ---------------------------------------------------------------------------
extern "C" void kernel_launch(void* const* d_in, const int* in_sizes, int n_in,
                              void* d_out, int out_size, void* d_ws, size_t ws_size,
                              hipStream_t stream) {
  (void)in_sizes; (void)n_in; (void)out_size; (void)ws_size;
  const float* x    = (const float*)d_in[0];  // (B,L,E)
  const float* rw   = (const float*)d_in[1];  // (NE,E)
  const float* rb   = (const float*)d_in[2];  // (NE,)
  const float* gate = (const float*)d_in[3];  // (NE,F,E)
  const float* enc  = (const float*)d_in[4];  // (NE,F,E)
  const float* dec  = (const float*)d_in[5];  // (NE,E,F)
  float* out = (float*)d_out;                 // T*E floats + 1 loss scalar

  char* ws = (char*)d_ws;
  size_t off = 0;
  auto bump = [&](size_t bytes) -> void* {
    void* p = ws + off;
    off = (off + bytes + 255) & ~(size_t)255;
    return p;
  };
  int*    eidx  = (int*)bump((size_t)RSLOT * 4);
  float*  eval  = (float*)bump((size_t)RSLOT * 4);
  int*    ranks = (int*)bump((size_t)RSLOT * 4);
  float*  imp   = (float*)bump((size_t)1024 * NE * 4);
  bf16_t* xd    = (bf16_t*)bump((size_t)NE * CAPACITY * E_DIM * 2);   //  32 MB
  bf16_t* h1    = (bf16_t*)bump((size_t)NE * CAPACITY * FF_DIM * 2);  // 128 MB
  float*  out2  = (float*)bump((size_t)NE * CAPACITY * E_DIM * 4);    //  64 MB

  moe_router_kernel<<<T_TOK / 8, 256, 0, stream>>>(x, rw, rb, eidx, eval, imp);
  moe_scan_kernel<<<1, 1024, 0, stream>>>(eidx, ranks, imp, out + (size_t)T_TOK * E_DIM);
  moe_dispatch_kernel<<<RSLOT, 256, 0, stream>>>(x, eidx, ranks, xd);

  dim3 g1(FF_DIM / 128, CAPACITY / 128, NE);  // 32 x 16 x 8
  moe_gemm_kernel<true><<<g1, 256, 0, stream>>>(xd, gate, enc, h1, nullptr,
                                                CAPACITY, FF_DIM, E_DIM);
  dim3 g2(E_DIM / 128, CAPACITY / 128, NE);   // 8 x 16 x 8
  moe_gemm_kernel<false><<<g2, 256, 0, stream>>>(h1, dec, nullptr, nullptr, out2,
                                                 CAPACITY, E_DIM, FF_DIM);

  moe_combine_kernel<<<T_TOK, 256, 0, stream>>>(out2, eidx, ranks, eval, out);
}
